// MultiLayerHORN_24309514895363
// MI455X (gfx1250) — compile-verified, matrix operand order checked by reading, not run on hardware
//
#include <hip/hip_runtime.h>
#include <hip/hip_bf16.h>

#define N_HID   128
#define T_STEPS 784
#define B_TOT   512
#define TILE_M  16
#define WAVES   8

typedef __attribute__((ext_vector_type(16))) __bf16 v16bf;
typedef __attribute__((ext_vector_type(8)))  __bf16 v8bf;
typedef __attribute__((ext_vector_type(8)))  float  v8f;

// float -> bf16: native fptrunc (RTNE), lowers to the hardware bf16 convert.
static __device__ __forceinline__ __bf16 f2bf(float f) { return (__bf16)f; }

// Branch-free tanh: sign(x) * (1 - e^{-2|x|}) / (1 + e^{-2|x|})
// using the native v_exp_f32 (2^x) and v_rcp_f32.  ~7 straight-line VALU ops,
// no exec-mask divergence on the recurrent critical path.
static __device__ __forceinline__ float fast_tanh(float x) {
    const float NLOG2E_2 = -2.8853900817779268f;   // -2*log2(e)
    float ax = __builtin_fabsf(x);
#if __has_builtin(__builtin_amdgcn_exp2f)
    float t = __builtin_amdgcn_exp2f(ax * NLOG2E_2);   // e^{-2|x|}
#else
    float t = exp2f(ax * NLOG2E_2);
#endif
#if __has_builtin(__builtin_amdgcn_rcpf)
    float r = __builtin_amdgcn_rcpf(1.0f + t);
#else
    float r = 1.0f / (1.0f + t);
#endif
    float th = (1.0f - t) * r;
    return __builtin_copysignf(th, x);
}

// K index inside a 32-wide K tile for the 16-bit WMMA operand register layout:
// lane group 0 (lanes 0-15): elems 0-7 -> K 0-7,  elems 8-15 -> K 16-23
// lane group 1 (lanes 16-31): elems 0-7 -> K 8-15, elems 8-15 -> K 24-31
static __device__ __forceinline__ int klocal(int j, int hi) {
    return hi * 8 + j + ((j >= 8) ? 8 : 0);
}

__global__ __launch_bounds__(256, 1)
void horn_kernel(const float* __restrict__ seq,   // [B, T]
                 const float* __restrict__ Wih1,  // [N1, 1]
                 const float* __restrict__ bih1,  // [N1]
                 const float* __restrict__ Whh1,  // [N1, N1]
                 const float* __restrict__ bhh1,  // [N1]
                 const float* __restrict__ Wih2,  // [N2, N1]
                 const float* __restrict__ bih2,  // [N2]
                 const float* __restrict__ Whh2,  // [N2, N2]
                 const float* __restrict__ bhh2,  // [N2]
                 const float* __restrict__ Wro,   // [10, N2]
                 const float* __restrict__ bro,   // [10]
                 float* __restrict__ out)         // [B, 10]
{
    __shared__ __attribute__((aligned(16))) __bf16 y1buf[2][TILE_M][N_HID];
    __shared__ __attribute__((aligned(16))) __bf16 y2buf[2][TILE_M][N_HID];
    __shared__ __attribute__((aligned(16))) __bf16 x1buf[TILE_M][N_HID];
    __shared__ float s_lds[TILE_M];
    __shared__ float xout[TILE_M][N_HID];

    const int tid  = threadIdx.x;
    const int wave = tid >> 5;
    const int lane = tid & 31;
    const int hi   = lane >> 4;    // lane half selects M/K sub-range
    const int lm   = lane & 15;
    const int n    = wave * 16 + lm;       // hidden column owned by this lane
    const int b0   = blockIdx.x * TILE_M;  // batch tile base

    const float inv_sqrt_n = 0.08838834764831845f; // 1/sqrt(128)

    // per-lane constants for the owned column
    const float w_in1 = Wih1[n];
    const float bias1 = bih1[n] + bhh1[n];
    const float bias2 = bih2[n] + bhh2[n];

    // ---- one-time: build bf16 B-operand fragments for the three weight
    // matrices (B[k][n] = W[n][k] since GEMMs are  h @ W^T).  Kept in VGPRs
    // for the whole T loop: zero per-step weight traffic. ----
    v16bf wB_hh1[4], wB_ih2[4], wB_hh2[4];
#pragma unroll
    for (int ko = 0; ko < 4; ++ko) {
        v16bf a, b, c;
#pragma unroll
        for (int j = 0; j < 16; ++j) {
            const int k = ko * 32 + klocal(j, hi);
            a[j] = f2bf(Whh1[n * N_HID + k]);
            b[j] = f2bf(Wih2[n * N_HID + k]);
            c[j] = f2bf(Whh2[n * N_HID + k]);
        }
        wB_hh1[ko] = a; wB_ih2[ko] = b; wB_hh2[ko] = c;
    }

    // oscillator states in WMMA C/D layout: acc[r] <-> (m = r + 8*hi, col n)
    v8f x1 = {}, y1 = {}, x2 = {}, y2 = {};

    // init LDS state buffers (zeros) + first input sample
#pragma unroll
    for (int r = 0; r < 8; ++r) {
        const int m = r + 8 * hi;
        y1buf[0][m][n] = f2bf(0.0f);
        y2buf[0][m][n] = f2bf(0.0f);
    }
    if (wave == 0 && lane < TILE_M)
        s_lds[lane] = seq[(b0 + lane) * T_STEPS + 0];
    __syncthreads();

    // A-fragment loader: row-major LDS tile -> WMMA A layout (two 16B DS loads)
    auto loadA = [&](const __bf16 (*buf)[N_HID], int ko) -> v16bf {
        const __bf16* row = buf[lm];
        v8bf lo = *(const v8bf*)(row + ko * 32 + hi * 8);
        v8bf hp = *(const v8bf*)(row + ko * 32 + 16 + hi * 8);
        return __builtin_shufflevector(lo, hp, 0, 1, 2, 3, 4, 5, 6, 7,
                                               8, 9, 10, 11, 12, 13, 14, 15);
    };

    for (int t = 0; t < T_STEPS; ++t) {
        const int p = t & 1;

        // ---------------- layer 1:  I1 = s*Wih1 + b + y1 @ Whh1^T ----------
        v8f acc;
#pragma unroll
        for (int r = 0; r < 8; ++r)
            acc[r] = s_lds[r + 8 * hi] * w_in1 + bias1;

        // issue all 4 A-fragment loads first so LDS latency overlaps WMMAs
        v16bf a1[4];
#pragma unroll
        for (int ko = 0; ko < 4; ++ko)
            a1[ko] = loadA(y1buf[p], ko);
#pragma unroll
        for (int ko = 0; ko < 4; ++ko)
            acc = __builtin_amdgcn_wmma_f32_16x16x32_bf16(
                false, a1[ko], false, wB_hh1[ko], (short)0, acc, false, false);

#pragma unroll
        for (int r = 0; r < 8; ++r) {
            const float accel = fast_tanh(acc[r] * inv_sqrt_n)  // ALPHA = 1
                                - 0.2f * y1[r]                  // 2*GAMMA
                                - x1[r];                        // OMEGA^2 = 1
            x1[r] += y1[r];                                     // H = 1
            y1[r] += accel;
        }
#pragma unroll
        for (int r = 0; r < 8; ++r) {
            const int m = r + 8 * hi;
            x1buf[m][n]         = f2bf(x1[r]);   // feeds layer-2 GEMM this step
            y1buf[p ^ 1][m][n]  = f2bf(y1[r]);   // feeds layer-1 GEMM next step
        }
        __syncthreads();   // x1/y1 published; everyone done reading y1buf[p]

        // prefetch next input sample (consumed only after next barrier)
        if (wave == 0 && lane < TILE_M && (t + 1) < T_STEPS)
            s_lds[lane] = seq[(b0 + lane) * T_STEPS + (t + 1)];

        // ---------------- layer 2:  I2 = x1 @ Wih2^T + b + y2 @ Whh2^T -----
        v8f acc2;
#pragma unroll
        for (int r = 0; r < 8; ++r) acc2[r] = bias2;

        // issue all 8 A-fragment loads up front, then the 8-WMMA chain
        v16bf ax[4], ay[4];
#pragma unroll
        for (int ko = 0; ko < 4; ++ko)
            ax[ko] = loadA(x1buf, ko);
#pragma unroll
        for (int ko = 0; ko < 4; ++ko)
            ay[ko] = loadA(y2buf[p], ko);
#pragma unroll
        for (int ko = 0; ko < 4; ++ko)
            acc2 = __builtin_amdgcn_wmma_f32_16x16x32_bf16(
                false, ax[ko], false, wB_ih2[ko], (short)0, acc2, false, false);
#pragma unroll
        for (int ko = 0; ko < 4; ++ko)
            acc2 = __builtin_amdgcn_wmma_f32_16x16x32_bf16(
                false, ay[ko], false, wB_hh2[ko], (short)0, acc2, false, false);

#pragma unroll
        for (int r = 0; r < 8; ++r) {
            const float accel = fast_tanh(acc2[r] * inv_sqrt_n)
                                - 0.2f * y2[r]
                                - x2[r];
            x2[r] += y2[r];
            y2[r] += accel;
        }
#pragma unroll
        for (int r = 0; r < 8; ++r)
            y2buf[p ^ 1][r + 8 * hi][n] = f2bf(y2[r]);
        __syncthreads();   // y2 published; everyone done reading y2buf[p]/x1buf
    }

    // ---------------- readout: out = x2 @ Wro^T + bro ----------------------
#pragma unroll
    for (int r = 0; r < 8; ++r)
        xout[r + 8 * hi][n] = x2[r];
    __syncthreads();

    if (tid < TILE_M * 10) {
        const int b = tid / 10, o = tid % 10;
        float sum = bro[o];
        for (int k = 0; k < N_HID; ++k)
            sum += xout[b][k] * Wro[o * N_HID + k];
        out[(b0 + b) * 10 + o] = sum;
    }
}

extern "C" void kernel_launch(void* const* d_in, const int* in_sizes, int n_in,
                              void* d_out, int out_size, void* d_ws, size_t ws_size,
                              hipStream_t stream) {
    const float* seq  = (const float*)d_in[0];
    const float* Wih1 = (const float*)d_in[1];
    const float* bih1 = (const float*)d_in[2];
    const float* Whh1 = (const float*)d_in[3];
    const float* bhh1 = (const float*)d_in[4];
    const float* Wih2 = (const float*)d_in[5];
    const float* bih2 = (const float*)d_in[6];
    const float* Whh2 = (const float*)d_in[7];
    const float* bhh2 = (const float*)d_in[8];
    const float* Wro  = (const float*)d_in[9];
    const float* bro  = (const float*)d_in[10];
    float* out = (float*)d_out;

    dim3 grid(B_TOT / TILE_M);   // 32 workgroups, one 16-row batch tile each
    dim3 block(WAVES * 32);      // 8 wave32s; each owns 16 hidden columns
    hipLaunchKernelGGL(horn_kernel, grid, block, 0, stream,
                       seq, Wih1, bih1, Whh1, bhh1,
                       Wih2, bih2, Whh2, bhh2, Wro, bro, out);
}
